// WaveRNN_11493332484479
// MI455X (gfx1250) — matrix operand order repaired and checked by reading, not compiled, not run on hardware
//
#include <hip/hip_runtime.h>
#include <hip/hip_bf16.h>
#include <cstddef>

// ---------------------------------------------------------------------------
// WaveRNN vocoder for MI455X (gfx1250, wave32, WMMA).
//
// Roofline: ~240 GFLOP total, ~1 GB of HBM traffic (~45us @ 23.3 TB/s) -> the
// binding constraint is the 16,000-step sequential GRU recurrence. The scan
// kernel keeps whh slices resident in LDS (192KB of the 320KB WGP LDS),
// carries h in registers, exchanges h through L2 with an agent-scope
// generation barrier, and runs the per-step matmul as 4 interleaved WMMA
// accumulator chains to minimize WMMA->WMMA hazard latency on the critical
// path. Parallel GEMMs use double-buffered global loads feeding
// v_wmma_f32_16x16x32_bf16 (f32 accumulation).
// ---------------------------------------------------------------------------

typedef __attribute__((ext_vector_type(16))) __bf16 v16bf;
typedef __attribute__((ext_vector_type(8)))  __bf16 v8bf;
typedef __attribute__((ext_vector_type(8)))  float  v8f;

#define B_SZ   8
#define T_STEPS 8000
#define HID    512
#define NGATE  1536
#define NWG    8          // workgroups in the persistent scan (must co-reside)
#define FEATK  128        // 102 features padded to 128 for K%32==0

__device__ __forceinline__ v16bf ld16(const __bf16* p0, const __bf16* p1) {
    v8bf lo = *(const v8bf*)p0;
    v8bf hi = *(const v8bf*)p1;
    v16bf r;
#pragma unroll
    for (int i = 0; i < 8; ++i) { r[i] = lo[i]; r[i + 8] = hi[i]; }
    return r;
}

// A fragment (16x32, M x K): lanes 0-15 rows, K chunks {0..7,16..23}; lanes
// 16-31 same rows, K chunks {8..15,24..31}.   (ISA 7.12.2, 16-bit A)
__device__ __forceinline__ v16bf load_afrag(const __bf16* A, int lda, int mbase,
                                            int k0, int lane) {
    int m  = mbase + (lane & 15);
    int ko = k0 + ((lane >> 4) << 3);
    const __bf16* r = A + (size_t)m * lda + ko;
    return ld16(r, r + 16);
}

// B fragment (32x16, K x N) from weights stored N-major (N x K row-major):
// lanes 0-15 columns with K {0..7,16..23}, lanes 16-31 same columns K {8..15,24..31}.
__device__ __forceinline__ v16bf load_bfrag(const __bf16* W, int ldw, int nbase,
                                            int k0, int lane) {
    int n  = nbase + (lane & 15);
    int ko = k0 + ((lane >> 4) << 3);
    const __bf16* r = W + (size_t)n * ldw + ko;
    return ld16(r, r + 16);
}

__device__ __forceinline__ v8f wmma_bf16(v16bf a, v16bf b, v8f c) {
    return __builtin_amdgcn_wmma_f32_16x16x32_bf16(false, a, false, b,
                                                   (short)0, c, false, false);
}

__device__ __forceinline__ float sigmoid_f(float x) {
    return 1.f / (1.f + expf(-x));
}

// ---------------------------------------------------------------------------
// Conv1d (k=7, same-pad, tanh).  in: (B, Cin, 800)  w: (20, Cin, 7)
// ---------------------------------------------------------------------------
__global__ void conv_tanh_kernel(const float* __restrict__ in,
                                 const float* __restrict__ w,
                                 const float* __restrict__ bias,
                                 float* __restrict__ out, int Cin) {
    int t = blockIdx.x * blockDim.x + threadIdx.x;
    int b = blockIdx.y;
    if (t >= 800) return;
    for (int c = 0; c < 20; ++c) {
        float acc = bias[c];
        for (int ci = 0; ci < Cin; ++ci) {
            const float* xr = in + ((size_t)b * Cin + ci) * 800;
            const float* wr = w + ((size_t)c * Cin + ci) * 7;
#pragma unroll
            for (int k = 0; k < 7; ++k) {
                int tt = t + k - 3;
                if (tt >= 0 && tt < 800) acc += wr[k] * xr[tt];
            }
        }
        out[((size_t)b * 20 + c) * 800 + t] = tanhf(acc);
    }
}

// ---------------------------------------------------------------------------
// Feature assembly: feat[bt][0:80]=up_mel, [80:100]=up_x, [100]=interp,
// [101]=gs_x, [102:128]=0.  Stored bf16, lda=128.
// ---------------------------------------------------------------------------
__global__ void feat_kernel(const float* __restrict__ mel,
                            const float* __restrict__ x_low,
                            const float* __restrict__ gs_x,
                            const float* __restrict__ conv2,
                            __bf16* __restrict__ feat) {
    long idx = (long)blockIdx.x * blockDim.x + threadIdx.x;
    if (idx >= (long)B_SZ * T_STEPS * FEATK) return;
    int  k  = (int)(idx & (FEATK - 1));
    long bt = idx >> 7;
    int  b  = (int)(bt / T_STEPS);
    int  t  = (int)(bt % T_STEPS);
    float v = 0.f;
    if (k < 80) {
        v = mel[((size_t)b * 80 + t / 100) * 80 + k];
    } else if (k < 100) {
        v = conv2[((size_t)b * 20 + (k - 80)) * 800 + t / 10];
    } else if (k == 100) {
        float src = (t + 0.5f) * 0.1f - 0.5f;
        src = fminf(fmaxf(src, 0.f), 799.f);
        int i0 = (int)floorf(src);
        int i1 = min(i0 + 1, 799);
        float w = src - (float)i0;
        v = x_low[(size_t)b * 800 + i0] * (1.f - w) +
            x_low[(size_t)b * 800 + i1] * w;
    } else if (k == 101) {
        v = gs_x[(size_t)b * T_STEPS + t];
    }
    feat[idx] = (__bf16)v;
}

// ---------------------------------------------------------------------------
// f32 -> bf16 cast with K-padding (src: N x K, dst: N x Kp, pad zeros)
// ---------------------------------------------------------------------------
__global__ void cast_pad_kernel(const float* __restrict__ src,
                                __bf16* __restrict__ dst, int N, int K, int Kp) {
    long idx = (long)blockIdx.x * blockDim.x + threadIdx.x;
    if (idx >= (long)N * Kp) return;
    int  k = (int)(idx % Kp);
    long n = idx / Kp;
    dst[idx] = (__bf16)((k < K) ? src[n * K + k] : 0.f);
}

// ---------------------------------------------------------------------------
// WMMA GEMM:  out(MxN) = A(MxK bf16) @ W(NxK bf16)^T + bias
//   mode 0: f32 out      mode 1: tanh -> bf16 out
// Block = 256 threads (8 waves), tile = 32(M) x 256(N), wave tile 16x64.
// Double-buffered operand fetch: next k-slice is issued before the current
// four WMMAs so s_wait_loadcnt overlaps with matrix math.
// ---------------------------------------------------------------------------
__global__ void gemm_bf16_kernel(const __bf16* __restrict__ A,
                                 const __bf16* __restrict__ W,
                                 const float* __restrict__ bias,
                                 void* __restrict__ out,
                                 int M, int N, int K, int mode) {
    const int lane  = threadIdx.x & 31;
    const int wv    = threadIdx.x >> 5;                 // 0..7
    const int mbase = blockIdx.x * 32 + (wv & 1) * 16;
    const int nbase = blockIdx.y * 256 + (wv >> 1) * 64;

    v8f acc[4] = {};
    v16bf a  = load_afrag(A, K, mbase, 0, lane);
    v16bf b0 = load_bfrag(W, K, nbase +  0, 0, lane);
    v16bf b1 = load_bfrag(W, K, nbase + 16, 0, lane);
    v16bf b2 = load_bfrag(W, K, nbase + 32, 0, lane);
    v16bf b3 = load_bfrag(W, K, nbase + 48, 0, lane);

    for (int k0 = 0; k0 < K; k0 += 32) {
        // branchless prefetch of the next slice (wraps to 0 on last iter)
        const int kn = (k0 + 32 < K) ? (k0 + 32) : 0;
        v16bf an  = load_afrag(A, K, mbase, kn, lane);
        v16bf bn0 = load_bfrag(W, K, nbase +  0, kn, lane);
        v16bf bn1 = load_bfrag(W, K, nbase + 16, kn, lane);
        v16bf bn2 = load_bfrag(W, K, nbase + 32, kn, lane);
        v16bf bn3 = load_bfrag(W, K, nbase + 48, kn, lane);

        acc[0] = wmma_bf16(a, b0, acc[0]);
        acc[1] = wmma_bf16(a, b1, acc[1]);
        acc[2] = wmma_bf16(a, b2, acc[2]);
        acc[3] = wmma_bf16(a, b3, acc[3]);

        a = an; b0 = bn0; b1 = bn1; b2 = bn2; b3 = bn3;
    }

    // C/D layout: vgpr r, lanes 0-15 -> M=r, lanes 16-31 -> M=r+8; lane&15 = N
    const int mrow = (lane >> 4) * 8;
    const int ncol = lane & 15;
#pragma unroll
    for (int j = 0; j < 4; ++j) {
        int n = nbase + j * 16 + ncol;
        float bv = bias[n];
#pragma unroll
        for (int r = 0; r < 8; ++r) {
            int m = mbase + mrow + r;
            float v = acc[j][r] + bv;
            if (mode == 0)
                ((float*)out)[(size_t)m * N + n] = v;
            else
                ((__bf16*)out)[(size_t)m * N + n] = (__bf16)tanhf(v);
        }
    }
}

// ---------------------------------------------------------------------------
// Persistent GRU scan. 8 WGs x 512 threads. WG w owns h columns
// [w*64, w*64+64) and the 192 matching rows of whh (r/z/n slices), held in
// LDS as bf16 with row pad 520 (bank-conflict-free ds_read_b128).
// Per step: 12 waves each run a K=512 reduction as FOUR interleaved WMMA
// accumulator chains (minimizes RAW hazard latency on the sequential critical
// path), gates fused in f32, h exchanged through L2 + agent-scope barrier.
// ---------------------------------------------------------------------------
#define WHP 520   // padded row length (bf16 elems) for whh_s and hA
#define HGP 17    // padded row length (f32) for the gate result tile

__global__ void gru_scan_kernel(const float* __restrict__ xg,
                                const __bf16* __restrict__ whh,
                                const float* __restrict__ bhh,
                                __bf16* __restrict__ hseq,
                                __bf16* __restrict__ hcur,
                                unsigned* sync_cnt, unsigned* sync_gen) {
    extern __shared__ unsigned char smem[];
    __bf16* whh_s = (__bf16*)smem;                              // [192][WHP]
    __bf16* hA    = (__bf16*)(smem + 192 * WHP * 2);            // [16][WHP]
    float*  hg    = (float*)(smem + 192 * WHP * 2 + 16 * WHP * 2); // [192][HGP]

    const int tid  = threadIdx.x;
    const int wg   = blockIdx.x;
    const int lane = tid & 31;
    const int wv   = tid >> 5;

    // Stage this WG's 192x512 whh slice into LDS (rows: r, z, n gate slices).
    for (int idx = tid; idx < 192 * 256; idx += 512) {
        int i = idx >> 8, c = idx & 255;
        int gate = i >> 6, jl = i & 63;
        int grow = gate * 512 + wg * 64 + jl;
        ((unsigned*)whh_s)[i * (WHP / 2) + c] =
            ((const unsigned*)whh)[(size_t)grow * 256 + c];
    }
    // h_0 = 0 (rows 8..15 of hA stay zero forever -> batch padding).
    for (int idx = tid; idx < 16 * (WHP / 2); idx += 512)
        ((unsigned*)hA)[idx] = 0u;

    const int b  = tid >> 6;          // batch row 0..7
    const int jl = tid & 63;          // local column
    const int jg = wg * 64 + jl;      // global h column
    float h_old = 0.f;
    const float br = bhh[jg], bz = bhh[512 + jg], bn = bhh[1024 + jg];

    // Per-wave fragment base pointers (constant across steps).
    const __bf16* pa = hA + (lane & 15) * WHP + ((lane >> 4) << 3);
    const __bf16* pb = whh_s + ((wv & 15) * 16 + (lane & 15)) * WHP +
                       ((lane >> 4) << 3);
    __syncthreads();

    for (int t = 0; t < T_STEPS; ++t) {
        // ---- hg = h @ whh_slice^T : 12 waves, 4 interleaved WMMA chains ----
        if (wv < 12) {
            v8f acc0 = {}, acc1 = {}, acc2 = {}, acc3 = {};
#pragma unroll
            for (int kt = 0; kt < 16; kt += 4) {
                const __bf16* pa0 = pa + (kt + 0) * 32;
                const __bf16* pa1 = pa + (kt + 1) * 32;
                const __bf16* pa2 = pa + (kt + 2) * 32;
                const __bf16* pa3 = pa + (kt + 3) * 32;
                const __bf16* pb0 = pb + (kt + 0) * 32;
                const __bf16* pb1 = pb + (kt + 1) * 32;
                const __bf16* pb2 = pb + (kt + 2) * 32;
                const __bf16* pb3 = pb + (kt + 3) * 32;
                v16bf a0 = ld16(pa0, pa0 + 16), w0 = ld16(pb0, pb0 + 16);
                v16bf a1 = ld16(pa1, pa1 + 16), w1 = ld16(pb1, pb1 + 16);
                v16bf a2 = ld16(pa2, pa2 + 16), w2 = ld16(pb2, pb2 + 16);
                v16bf a3 = ld16(pa3, pa3 + 16), w3 = ld16(pb3, pb3 + 16);
                acc0 = wmma_bf16(a0, w0, acc0);
                acc1 = wmma_bf16(a1, w1, acc1);
                acc2 = wmma_bf16(a2, w2, acc2);
                acc3 = wmma_bf16(a3, w3, acc3);
            }
            v8f accs = (acc0 + acc1) + (acc2 + acc3);
            int n_l = wv * 16 + (lane & 15);
            int mr  = (lane >> 4) * 8;
#pragma unroll
            for (int r = 0; r < 8; ++r) hg[n_l * HGP + mr + r] = accs[r];
        }
        __syncthreads();

        // ---- gate fusion (f32), h carried in a register ----
        size_t xbase = ((size_t)b * T_STEPS + t) * NGATE;
        float xr = xg[xbase + jg];
        float xz = xg[xbase + 512 + jg];
        float xn = xg[xbase + 1024 + jg];
        float hr = hg[jl * HGP + b] + br;
        float hz = hg[(64 + jl) * HGP + b] + bz;
        float hn = hg[(128 + jl) * HGP + b] + bn;
        float rg = sigmoid_f(xr + hr);
        float zg = sigmoid_f(xz + hz);
        float ng = tanhf(xn + rg * hn);
        float hnew = (1.f - zg) * ng + zg * h_old;
        h_old = hnew;
        hseq[((size_t)b * T_STEPS + t) * HID + jg] = (__bf16)hnew;
        hcur[b * HID + jg] = (__bf16)hnew;
        if (t + 1 < T_STEPS) {
            __builtin_prefetch(&xg[xbase + NGATE + jg], 0, 1);        // r slice
            __builtin_prefetch(&xg[xbase + NGATE + 1024 + jg], 0, 1); // n slice
        }

        // ---- device-wide barrier (8 co-resident WGs) ----
        __threadfence();            // release: per-wave store flush to L2
        __syncthreads();
        if (tid == 0) {
            unsigned g = __hip_atomic_load(sync_gen, __ATOMIC_RELAXED,
                                           __HIP_MEMORY_SCOPE_AGENT);
            unsigned a = __hip_atomic_fetch_add(sync_cnt, 1u, __ATOMIC_ACQ_REL,
                                                __HIP_MEMORY_SCOPE_AGENT);
            if (a == NWG - 1) {
                __hip_atomic_store(sync_cnt, 0u, __ATOMIC_RELAXED,
                                   __HIP_MEMORY_SCOPE_AGENT);
                __hip_atomic_fetch_add(sync_gen, 1u, __ATOMIC_RELEASE,
                                       __HIP_MEMORY_SCOPE_AGENT);
            } else {
                while (__hip_atomic_load(sync_gen, __ATOMIC_ACQUIRE,
                                         __HIP_MEMORY_SCOPE_AGENT) == g)
                    __builtin_amdgcn_s_sleep(2);
            }
        }
        __syncthreads();
        __threadfence();            // acquire: invalidate near caches

        // ---- refresh full h (bf16) into LDS A-operand buffer ----
        for (int idx = tid; idx < 8 * 256; idx += 512) {
            int m = idx >> 8, c = idx & 255;
            ((unsigned*)hA)[m * (WHP / 2) + c] = ((const unsigned*)hcur)[m * 256 + c];
        }
        __syncthreads();
    }
}

// ---------------------------------------------------------------------------
// Output head: out[bt][n] = out_b[n] + pre[bt] . out_w[n]   (N=30, K=256)
// ---------------------------------------------------------------------------
__global__ void out_proj_kernel(const __bf16* __restrict__ pre,
                                const float* __restrict__ out_w,
                                const float* __restrict__ out_b,
                                float* __restrict__ out) {
    int bt = blockIdx.x * blockDim.x + threadIdx.x;
    if (bt >= B_SZ * T_STEPS) return;
    float acc[30];
#pragma unroll
    for (int n = 0; n < 30; ++n) acc[n] = out_b[n];
    const __bf16* p = pre + (size_t)bt * 256;
    for (int k = 0; k < 256; k += 8) {
        float pv[8];
#pragma unroll
        for (int i = 0; i < 8; ++i) pv[i] = (float)p[k + i];
#pragma unroll
        for (int n = 0; n < 30; ++n) {
            float a = acc[n];
#pragma unroll
            for (int i = 0; i < 8; ++i) a += pv[i] * out_w[n * 256 + k + i];
            acc[n] = a;
        }
    }
#pragma unroll
    for (int n = 0; n < 30; ++n) out[(size_t)bt * 30 + n] = acc[n];
}

// ---------------------------------------------------------------------------
extern "C" void kernel_launch(void* const* d_in, const int* in_sizes, int n_in,
                              void* d_out, int out_size, void* d_ws, size_t ws_size,
                              hipStream_t stream) {
    const float* mel     = (const float*)d_in[0];
    const float* x_low   = (const float*)d_in[1];
    const float* gs_x    = (const float*)d_in[2];
    const float* conv_w0 = (const float*)d_in[3];
    const float* conv_b0 = (const float*)d_in[4];
    const float* conv_w1 = (const float*)d_in[5];
    const float* conv_b1 = (const float*)d_in[6];
    const float* conv_w2 = (const float*)d_in[7];
    const float* conv_b2 = (const float*)d_in[8];
    const float* wih0    = (const float*)d_in[9];
    const float* whh0    = (const float*)d_in[10];
    const float* bih0    = (const float*)d_in[11];
    const float* bhh0    = (const float*)d_in[12];
    const float* wih1    = (const float*)d_in[13];
    const float* whh1    = (const float*)d_in[14];
    const float* bih1    = (const float*)d_in[15];
    const float* bhh1    = (const float*)d_in[16];
    const float* pre_w   = (const float*)d_in[17];
    const float* pre_b   = (const float*)d_in[18];
    const float* out_w   = (const float*)d_in[19];
    const float* out_b   = (const float*)d_in[20];

    char* ws = (char*)d_ws;
    size_t off = 0;
    auto bump = [&](size_t bytes) {
        size_t o = off;
        off = (off + bytes + 255) & ~(size_t)255;
        return o;
    };
    const size_t M = (size_t)B_SZ * T_STEPS;  // 64000

    size_t o_sync  = bump(256);
    size_t o_convA = bump((size_t)B_SZ * 20 * 800 * 4);
    size_t o_convB = bump((size_t)B_SZ * 20 * 800 * 4);
    size_t o_feat  = bump(M * FEATK * 2);
    size_t o_wih0  = bump((size_t)NGATE * FEATK * 2);
    size_t o_whh0  = bump((size_t)NGATE * HID * 2);
    size_t o_wih1  = bump((size_t)NGATE * HID * 2);
    size_t o_whh1  = bump((size_t)NGATE * HID * 2);
    size_t o_prew  = bump((size_t)256 * HID * 2);
    size_t o_pre   = bump(M * 256 * 2);
    size_t o_hseq  = bump(M * HID * 2);
    size_t o_hcur  = bump((size_t)16 * HID * 2);
    size_t o_xg    = bump(M * NGATE * 4);
    (void)ws_size; (void)in_sizes; (void)n_in; (void)out_size;

    unsigned* sync_cnt = (unsigned*)(ws + o_sync);
    unsigned* sync_gen = (unsigned*)(ws + o_sync + 64);
    float*  convA = (float*)(ws + o_convA);
    float*  convB = (float*)(ws + o_convB);
    __bf16* featA = (__bf16*)(ws + o_feat);
    __bf16* wih0b = (__bf16*)(ws + o_wih0);
    __bf16* whh0b = (__bf16*)(ws + o_whh0);
    __bf16* wih1b = (__bf16*)(ws + o_wih1);
    __bf16* whh1b = (__bf16*)(ws + o_whh1);
    __bf16* prewb = (__bf16*)(ws + o_prew);
    __bf16* preb  = (__bf16*)(ws + o_pre);
    __bf16* hseq  = (__bf16*)(ws + o_hseq);
    __bf16* hcur  = (__bf16*)(ws + o_hcur);
    float*  xg    = (float*)(ws + o_xg);

    hipMemsetAsync(ws + o_sync, 0, 256, stream);

    // Conv stack (B, C, 800)
    dim3 cgrid((800 + 127) / 128, B_SZ);
    conv_tanh_kernel<<<cgrid, 128, 0, stream>>>(x_low, conv_w0, conv_b0, convA, 1);
    conv_tanh_kernel<<<cgrid, 128, 0, stream>>>(convA, conv_w1, conv_b1, convB, 20);
    conv_tanh_kernel<<<cgrid, 128, 0, stream>>>(convB, conv_w2, conv_b2, convA, 20);

    // Feature tensor (bf16, M x 128)
    feat_kernel<<<(unsigned)((M * FEATK) / 256), 256, 0, stream>>>(
        mel, x_low, gs_x, convA, featA);

    // Weight casts (bf16, N x K row-major = WMMA B operand)
    cast_pad_kernel<<<(NGATE * FEATK + 255) / 256, 256, 0, stream>>>(wih0, wih0b, NGATE, 102, FEATK);
    cast_pad_kernel<<<(NGATE * HID + 255) / 256, 256, 0, stream>>>(whh0, whh0b, NGATE, HID, HID);
    cast_pad_kernel<<<(NGATE * HID + 255) / 256, 256, 0, stream>>>(wih1, wih1b, NGATE, HID, HID);
    cast_pad_kernel<<<(NGATE * HID + 255) / 256, 256, 0, stream>>>(whh1, whh1b, NGATE, HID, HID);
    cast_pad_kernel<<<(256 * HID + 255) / 256, 256, 0, stream>>>(pre_w, prewb, 256, HID, HID);

    const size_t scan_lds = 192 * WHP * 2 + 16 * WHP * 2 + 192 * HGP * 4;

    // ---- GRU layer 0 ----
    gemm_bf16_kernel<<<dim3((unsigned)(M / 32), NGATE / 256), 256, 0, stream>>>(
        featA, wih0b, bih0, xg, (int)M, NGATE, FEATK, 0);
    gru_scan_kernel<<<NWG, 512, scan_lds, stream>>>(
        xg, whh0b, bhh0, hseq, hcur, sync_cnt, sync_gen);

    // ---- GRU layer 1 ----
    gemm_bf16_kernel<<<dim3((unsigned)(M / 32), NGATE / 256), 256, 0, stream>>>(
        hseq, wih1b, bih1, xg, (int)M, NGATE, HID, 0);
    gru_scan_kernel<<<NWG, 512, scan_lds, stream>>>(
        xg, whh1b, bhh1, hseq, hcur, sync_cnt, sync_gen);

    // ---- prelinear (tanh -> bf16) + output head ----
    gemm_bf16_kernel<<<dim3((unsigned)(M / 32), 1), 256, 0, stream>>>(
        hseq, prewb, pre_b, preb, (int)M, 256, HID, 1);
    out_proj_kernel<<<(unsigned)((M + 127) / 128), 128, 0, stream>>>(
        preb, out_w, out_b, (float*)d_out);
}
